// IaKNN_42314017800873
// MI455X (gfx1250) — matrix operand order repaired and checked by reading, not compiled, not run on hardware
//
#include <hip/hip_runtime.h>
#include <hip/hip_bf16.h>
#include <math.h>

typedef __attribute__((ext_vector_type(16))) _Float16 v16h;
typedef __attribute__((ext_vector_type(8)))  _Float16 v8h;
typedef __attribute__((ext_vector_type(4)))  _Float16 v4h;
typedef __attribute__((ext_vector_type(8)))  float    v8f;

#define DT_C   0.1f
#define NIMG_C 51200
#define NB_C   512
#define GST    17   // gates LDS row stride (pad 16 -> 17 to kill bank conflicts)

// ---------- WMMA fragment loaders (contiguous k = 16h + s order, wave32) ----------

// A: 16x32 f16 tile from row-major matrix, lda in elements (lda % 8 == 0).
// lane<16:  row M=lane,    elems 0..15 = K 0..15
// lane>=16: row M=lane-16, elems 0..15 = K 16..31
// -> two contiguous 16-byte runs per lane: 2 x ds_load_b128.
__device__ __forceinline__ v16h load_a16(const _Float16* __restrict__ A, int lda) {
  int lane = threadIdx.x & 31;
  int m = lane & 15;
  int kb0 = (lane < 16) ? 0 : 16;
  const _Float16* p = A + m * lda + kb0;
  v8h lo = *(const v8h*)(p);
  v8h hi = *(const v8h*)(p + 8);
  return __builtin_shufflevector(lo, hi, 0, 1, 2, 3, 4, 5, 6, 7,
                                 8, 9, 10, 11, 12, 13, 14, 15);
}

// B: 32x16 tile staged lane-packed: BT[col][32] f16 (tile = 512 halves, 1KB).
// lane<16: col=lane, K=0..15 ; lane>=16: col=lane-16, K=16..31
// -> one contiguous 32-byte run per lane (32B aligned): 2 x ds_load_b128.
__device__ __forceinline__ v16h load_bT(const _Float16* __restrict__ BT) {
  int lane = threadIdx.x & 31;
  int n = lane & 15;
  int kb = (lane < 16) ? 0 : 16;
  return *(const v16h*)(BT + n * 32 + kb);
}

__device__ __forceinline__ float sigf(float x) { return 1.0f / (1.0f + __expf(-x)); }

// =====================================================================
// Kernel 1: per-frame CNN. One wave per image (8 images / 256-thr block).
// Writes flatten(128) f16 row-major: X1[img*128 + k].
// =====================================================================
__global__ void conv_feat_kernel(const float* __restrict__ obs,
                                 const float* __restrict__ c1w, const float* __restrict__ c1b,
                                 const float* __restrict__ c2w, const float* __restrict__ c2b,
                                 _Float16* __restrict__ X1) {
  __shared__ float w1[576], b1[16], w2[4608], b2[32];
  __shared__ float s_in[8][400];
  __shared__ float s_a1[8][400];
  int tid = threadIdx.x;
  for (int i = tid; i < 576;  i += 256) w1[i] = c1w[i];
  for (int i = tid; i < 16;   i += 256) b1[i] = c1b[i];
  for (int i = tid; i < 4608; i += 256) w2[i] = c2w[i];
  for (int i = tid; i < 32;   i += 256) b2[i] = c2b[i];

  int wave = tid >> 5, lane = tid & 31;
  int img = blockIdx.x * 8 + wave;
  // vectorized input load: 100 float4 per image
  const float4* src4 = (const float4*)(obs + (size_t)img * 400);
  float4* dst4 = (float4*)&s_in[wave][0];
  for (int i = lane; i < 100; i += 32) dst4[i] = src4[i];
  if (img + 8 < NIMG_C) __builtin_prefetch(obs + (size_t)(img + 8) * 400, 0, 0);
  __syncthreads();

  // conv1 (4->16, 3x3 SAME) + relu + 2x2 maxpool -> 16x5x5
  for (int idx = lane; idx < 400; idx += 32) {
    int oc = idx / 25, rem = idx % 25, py = rem / 5, px = rem % 5;
    float mx = 0.0f;
    for (int dy = 0; dy < 2; ++dy)
      for (int dx = 0; dx < 2; ++dx) {
        int y = py * 2 + dy, x = px * 2 + dx;
        float acc = b1[oc];
        for (int ic = 0; ic < 4; ++ic)
          for (int ky = 0; ky < 3; ++ky) {
            int iy = y + ky - 1; if (iy < 0 || iy >= 10) continue;
            for (int kx = 0; kx < 3; ++kx) {
              int ix = x + kx - 1; if (ix < 0 || ix >= 10) continue;
              acc += w1[((oc * 4 + ic) * 3 + ky) * 3 + kx] * s_in[wave][ic * 100 + iy * 10 + ix];
            }
          }
        float r = acc > 0.f ? acc : 0.f;
        mx = r > mx ? r : mx;
      }
    s_a1[wave][idx] = mx;
  }
  __syncthreads();

  // conv2 (16->32, 3x3 SAME on 5x5) + relu + 2x2 maxpool -> 32x2x2 = 128
  for (int idx = lane; idx < 128; idx += 32) {
    int oc = idx / 4, rem = idx % 4, py = rem / 2, px = rem % 2;
    float mx = 0.0f;
    for (int dy = 0; dy < 2; ++dy)
      for (int dx = 0; dx < 2; ++dx) {
        int y = py * 2 + dy, x = px * 2 + dx;
        float acc = b2[oc];
        for (int ic = 0; ic < 16; ++ic)
          for (int ky = 0; ky < 3; ++ky) {
            int iy = y + ky - 1; if (iy < 0 || iy >= 5) continue;
            for (int kx = 0; kx < 3; ++kx) {
              int ix = x + kx - 1; if (ix < 0 || ix >= 5) continue;
              acc += w2[((oc * 16 + ic) * 3 + ky) * 3 + kx] * s_a1[wave][ic * 25 + iy * 5 + ix];
            }
          }
        float r = acc > 0.f ? acc : 0.f;
        mx = r > mx ? r : mx;
      }
    X1[(size_t)img * 128 + idx] = (_Float16)mx;
  }
}

// =====================================================================
// Kernel 2: FC1(128->128)+BN+ReLU, FC2(128->64)+BN+ReLU via WMMA.
// 16 images per block; activations staged lane-packed for B fragments.
// Output: featT[t][b][64] f16 (8 contiguous f16 per lane -> b128 store).
// =====================================================================
__global__ void fc_kernel(const _Float16* __restrict__ X1,
                          const float* __restrict__ f1w, const float* __restrict__ f1b,
                          const float* __restrict__ bn1g, const float* __restrict__ bn1b,
                          const float* __restrict__ f2w, const float* __restrict__ f2b,
                          const float* __restrict__ bn2g, const float* __restrict__ bn2b,
                          _Float16* __restrict__ featT) {
  __shared__ __align__(32) _Float16 w1h[128 * 128];
  __shared__ __align__(32) _Float16 w2h[64 * 128];
  __shared__ __align__(32) _Float16 xbT[4 * 16 * 32];   // 4 K-tiles, lane-packed
  __shared__ __align__(32) _Float16 y1T[4 * 16 * 32];
  __shared__ float sb1[128], sg1[128], sbb1[128];
  __shared__ float sb2[64], sg2[64], sbb2[64];
  int tid = threadIdx.x;
  for (int i = tid; i < 128 * 128; i += 256) w1h[i] = (_Float16)f1w[i];
  for (int i = tid; i < 64 * 128;  i += 256) w2h[i] = (_Float16)f2w[i];
  for (int i = tid; i < 128; i += 256) { sb1[i] = f1b[i]; sg1[i] = bn1g[i]; sbb1[i] = bn1b[i]; }
  for (int i = tid; i < 64;  i += 256) { sb2[i] = f2b[i]; sg2[i] = bn2g[i]; sbb2[i] = bn2b[i]; }
  int n0 = blockIdx.x * 16;
  for (int i = tid; i < 2048; i += 256) {       // col-major-coalesced stage
    int col = i >> 7, k = i & 127;
    xbT[((k >> 5) * 16 + col) * 32 + (k & 31)] = X1[(size_t)(n0 + col) * 128 + k];
  }
  __syncthreads();

  int wave = tid >> 5, lane = tid & 31;
  int n = lane & 15, mb = (lane < 16) ? 0 : 8;
  const float inv_bn = rsqrtf(1.0f + 1e-5f);

  // FC1: 8 waves x (16 rows each), K = 128 in 4 steps
  {
    int m0 = wave * 16;
    v8f acc = {};
#pragma unroll
    for (int kt = 0; kt < 4; ++kt) {
      v16h a = load_a16(&w1h[m0 * 128 + kt * 32], 128);
      v16h b = load_bT(&xbT[kt * 512]);
      acc = __builtin_amdgcn_wmma_f32_16x16x32_f16(false, a, false, b, (short)0, acc, false, false);
    }
#pragma unroll
    for (int r = 0; r < 8; ++r) {
      int m = m0 + mb + r;
      float v = acc[r] + sb1[m];
      v = sg1[m] * v * inv_bn + sbb1[m];
      v = v > 0.f ? v : 0.f;
      y1T[((m >> 5) * 16 + n) * 32 + (m & 31)] = (_Float16)v;
    }
  }
  __syncthreads();

  // FC2: waves 0..3 (wave-uniform branch, EXEC all ones inside)
  if (wave < 4) {
    int m0 = wave * 16;
    v8f acc = {};
#pragma unroll
    for (int kt = 0; kt < 4; ++kt) {
      v16h a = load_a16(&w2h[m0 * 128 + kt * 32], 128);
      v16h b = load_bT(&y1T[kt * 512]);
      acc = __builtin_amdgcn_wmma_f32_16x16x32_f16(false, a, false, b, (short)0, acc, false, false);
    }
    int img = n0 + n;
    int bidx = img / 100, t = img % 100;
    _Float16* dst = &featT[((size_t)t * NB_C + bidx) * 64 + m0 + mb];
#pragma unroll
    for (int r = 0; r < 8; ++r) {
      int m = m0 + mb + r;
      float v = acc[r] + sb2[m];
      v = sg2[m] * v * inv_bn + sbb2[m];
      v = v > 0.f ? v : 0.f;
      dst[r] = (_Float16)v;   // 8 contiguous f16 -> merged wide store
    }
  }
}

// =====================================================================
// Kernel 3: encoder LSTM (H=32), 16 batch columns per block, T=100.
// Weight A-fragments hoisted into registers outside the time loop.
// =====================================================================
__global__ void enc_lstm_kernel(const _Float16* __restrict__ featT,
                                const float* __restrict__ eWih, const float* __restrict__ eWhh,
                                const float* __restrict__ ebih, const float* __restrict__ ebhh,
                                float* __restrict__ hfin, float* __restrict__ cfin) {
  __shared__ __align__(32) _Float16 wih[128 * 64];
  __shared__ __align__(32) _Float16 whh[128 * 32];
  __shared__ __align__(32) _Float16 xbT[2 * 16 * 32];   // 2 K-tiles lane-packed
  __shared__ __align__(32) _Float16 hbT[16 * 32];
  __shared__ float gates[128 * GST];
  __shared__ float hC[32 * 16], cC[32 * 16];
  __shared__ float bias[128];
  int tid = threadIdx.x;
  for (int i = tid; i < 128 * 64; i += 256) wih[i] = (_Float16)eWih[i];
  for (int i = tid; i < 128 * 32; i += 256) whh[i] = (_Float16)eWhh[i];
  for (int i = tid; i < 128; i += 256) bias[i] = ebih[i] + ebhh[i];
  for (int i = tid; i < 32 * 16; i += 256) { hC[i] = 0.f; cC[i] = 0.f; }
  for (int i = tid; i < 16 * 32; i += 256) hbT[i] = (_Float16)0.0f;
  int bc0 = blockIdx.x * 16;
  int wave = tid >> 5, lane = tid & 31;
  int n = lane & 15, mb = (lane < 16) ? 0 : 8;
  int m0 = wave * 16;
  __syncthreads();

  // loop-invariant weight fragments
  v16h aX0 = load_a16(&wih[m0 * 64 + 0],  64);
  v16h aX1 = load_a16(&wih[m0 * 64 + 32], 64);
  v16h aH  = load_a16(&whh[m0 * 32], 32);

  // per-thread staging coords: 4 contiguous f16 each
  int scol = tid >> 4, sk0 = (tid & 15) * 4;

  for (int t = 0; t < 100; ++t) {
    {
      v4h tmp = *(const v4h*)&featT[((size_t)t * NB_C + bc0 + scol) * 64 + sk0];
      *(v4h*)&xbT[((sk0 >> 5) * 16 + scol) * 32 + (sk0 & 31)] = tmp;
    }
    __syncthreads();
    v8f acc = {};
    acc = __builtin_amdgcn_wmma_f32_16x16x32_f16(false, aX0, false, load_bT(&xbT[0]),
                                                 (short)0, acc, false, false);
    acc = __builtin_amdgcn_wmma_f32_16x16x32_f16(false, aX1, false, load_bT(&xbT[512]),
                                                 (short)0, acc, false, false);
    acc = __builtin_amdgcn_wmma_f32_16x16x32_f16(false, aH,  false, load_bT(&hbT[0]),
                                                 (short)0, acc, false, false);
#pragma unroll
    for (int r = 0; r < 8; ++r) {
      int m = m0 + mb + r;
      gates[m * GST + n] = acc[r] + bias[m];
    }
    __syncthreads();
    for (int i = tid; i < 32 * 16; i += 256) {
      int u = i >> 4, col = i & 15;
      float ig = gates[u * GST + col];
      float fg = gates[(32 + u) * GST + col];
      float gg = gates[(64 + u) * GST + col];
      float og = gates[(96 + u) * GST + col];
      float c = sigf(fg) * cC[i] + sigf(ig) * tanhf(gg);
      float h = sigf(og) * tanhf(c);
      cC[i] = c; hC[i] = h;
      hbT[col * 32 + u] = (_Float16)h;
    }
    __syncthreads();
  }
  for (int i = tid; i < 32 * 16; i += 256) {
    int u = i >> 4, col = i & 15;
    hfin[(size_t)(bc0 + col) * 32 + u] = hC[i];
    cfin[(size_t)(bc0 + col) * 32 + u] = cC[i];
  }
}

// =====================================================================
// Kernel 4: decoder LSTM (5 steps) + output proj + physics integration.
// traj written f32 [b][s][24] and f16 [s][b][24].
// =====================================================================
__global__ void dec_kernel(const float* __restrict__ obs,
                           const float* __restrict__ dWih, const float* __restrict__ dWhh,
                           const float* __restrict__ dbih, const float* __restrict__ dbhh,
                           const float* __restrict__ ow, const float* __restrict__ ob,
                           const float* __restrict__ hfin, const float* __restrict__ cfin,
                           float* __restrict__ trajF, _Float16* __restrict__ trajH) {
  __shared__ __align__(32) _Float16 wih[128 * 32], whh[128 * 32], owp[16 * 32];
  __shared__ __align__(32) _Float16 xbT[16 * 32], hbT[16 * 32];
  __shared__ float gates[128 * GST];
  __shared__ float hC[32 * 16], cC[32 * 16];
  __shared__ float bias[128], ob_s[12];
  __shared__ float obs_s[16][4];
  __shared__ float accel[5][12 * 16];
  int tid = threadIdx.x;
  for (int i = tid; i < 128 * 32; i += 256) { wih[i] = (_Float16)dWih[i]; whh[i] = (_Float16)dWhh[i]; }
  for (int i = tid; i < 16 * 32; i += 256) {
    int m = i >> 5, k = i & 31;
    owp[i] = (m < 12) ? (_Float16)ow[m * 32 + k] : (_Float16)0.0f;
  }
  for (int i = tid; i < 128; i += 256) bias[i] = dbih[i] + dbhh[i];
  for (int i = tid; i < 12;  i += 256) ob_s[i] = ob[i];
  int bc0 = blockIdx.x * 16;
  for (int i = tid; i < 32 * 16; i += 256) {
    int u = i >> 4, col = i & 15;
    float h = hfin[(size_t)(bc0 + col) * 32 + u];
    float c = cfin[(size_t)(bc0 + col) * 32 + u];
    hC[i] = h; cC[i] = c;
    hbT[col * 32 + u] = (_Float16)h;
    xbT[col * 32 + u] = (_Float16)h;   // din = enc_out[:, -1]
  }
  if (tid < 64) {
    int col = tid >> 2, j = tid & 3;
    obs_s[col][j] = obs[((size_t)(bc0 + col) * 100 + 99) * 400 + j];
  }
  __syncthreads();

  int wave = tid >> 5, lane = tid & 31;
  int n = lane & 15, mb = (lane < 16) ? 0 : 8;
  int m0 = wave * 16;

  v16h aX = load_a16(&wih[m0 * 32], 32);
  v16h aH = load_a16(&whh[m0 * 32], 32);
  v16h aO = load_a16(&owp[0], 32);

  for (int s = 0; s < 5; ++s) {
    v8f acc = {};
    acc = __builtin_amdgcn_wmma_f32_16x16x32_f16(false, aX, false, load_bT(&xbT[0]),
                                                 (short)0, acc, false, false);
    acc = __builtin_amdgcn_wmma_f32_16x16x32_f16(false, aH, false, load_bT(&hbT[0]),
                                                 (short)0, acc, false, false);
#pragma unroll
    for (int r = 0; r < 8; ++r) {
      int m = m0 + mb + r;
      gates[m * GST + n] = acc[r] + bias[m];
    }
    __syncthreads();
    for (int i = tid; i < 32 * 16; i += 256) {
      int u = i >> 4, col = i & 15;
      float ig = gates[u * GST + col], fg = gates[(32 + u) * GST + col];
      float gg = gates[(64 + u) * GST + col], og = gates[(96 + u) * GST + col];
      float c = sigf(fg) * cC[i] + sigf(ig) * tanhf(gg);
      float h = sigf(og) * tanhf(c);
      cC[i] = c; hC[i] = h;
      hbT[col * 32 + u] = (_Float16)h;
    }
    __syncthreads();
    if (wave == 0) {   // accel = h @ ow.T + ob (M padded 12->16)
      v8f acc2 = {};
      acc2 = __builtin_amdgcn_wmma_f32_16x16x32_f16(false, aO, false, load_bT(&hbT[0]),
                                                    (short)0, acc2, false, false);
#pragma unroll
      for (int r = 0; r < 8; ++r) {
        int m = mb + r;
        if (m < 12) accel[s][m * 16 + n] = acc2[r] + ob_s[m];
      }
    }
    for (int i = tid; i < 16 * 32; i += 256) xbT[i] = hbT[i];  // din = dh
    __syncthreads();
  }

  // physics: v = v0 + DT*cumsum(a); disp = v*DT + 0.5*a*DT^2; p = p0 + cumsum(disp)
  if (tid < 16) {
    int col = tid, b = bc0 + col;
    float vx[6], vy[6], px[6], py[6];
    for (int nv = 0; nv < 6; ++nv) { vx[nv] = 0.f; vy[nv] = 0.f; px[nv] = 0.f; py[nv] = 0.f; }
    px[0] = obs_s[col][0]; py[0] = obs_s[col][1];
    vx[0] = obs_s[col][2]; vy[0] = obs_s[col][3];
    for (int s = 0; s < 5; ++s) {
      for (int nv = 0; nv < 6; ++nv) {
        float ax = accel[s][(nv * 2 + 0) * 16 + col];
        float ay = accel[s][(nv * 2 + 1) * 16 + col];
        vx[nv] += DT_C * ax; vy[nv] += DT_C * ay;
        px[nv] += vx[nv] * DT_C + 0.5f * ax * DT_C * DT_C;
        py[nv] += vy[nv] * DT_C + 0.5f * ay * DT_C * DT_C;
        float tv[4] = { px[nv], py[nv], vx[nv], vy[nv] };
        for (int c2 = 0; c2 < 4; ++c2) {
          int k = nv * 4 + c2;
          trajF[((size_t)b * 5 + s) * 24 + k] = tv[c2];
          trajH[((size_t)s * NB_C + b) * 24 + k] = (_Float16)tv[c2];
        }
      }
    }
  }
}

// =====================================================================
// Kernel 5: Q/R LSTM (input 24 padded to K=32) + |h @ fw.T + fb| diag out.
// Launched twice (Q weights, R weights).
// =====================================================================
__global__ void qr_lstm_kernel(const _Float16* __restrict__ trajH,
                               const float* __restrict__ Wih, const float* __restrict__ Whh,
                               const float* __restrict__ bih, const float* __restrict__ bhh,
                               const float* __restrict__ fw, const float* __restrict__ fb,
                               float* __restrict__ Dout) {
  __shared__ __align__(32) _Float16 wih[128 * 32], whh[128 * 32], fwp[32 * 32];
  __shared__ __align__(32) _Float16 xbT[16 * 32], hbT[16 * 32];
  __shared__ float gates[128 * GST];
  __shared__ float hC[32 * 16], cC[32 * 16];
  __shared__ float bias[128], fb_s[24];
  int tid = threadIdx.x;
  for (int i = tid; i < 128 * 32; i += 256) {
    int m = i >> 5, k = i & 31;
    wih[i] = (k < 24) ? (_Float16)Wih[m * 24 + k] : (_Float16)0.0f;
    whh[i] = (_Float16)Whh[i];
  }
  for (int i = tid; i < 32 * 32; i += 256) {
    int m = i >> 5, k = i & 31;
    fwp[i] = (m < 24) ? (_Float16)fw[m * 32 + k] : (_Float16)0.0f;
  }
  for (int i = tid; i < 128; i += 256) bias[i] = bih[i] + bhh[i];
  for (int i = tid; i < 24;  i += 256) fb_s[i] = fb[i];
  for (int i = tid; i < 32 * 16; i += 256) { hC[i] = 0.f; cC[i] = 0.f; }
  for (int i = tid; i < 16 * 32; i += 256) hbT[i] = (_Float16)0.0f;
  int bc0 = blockIdx.x * 16;
  int wave = tid >> 5, lane = tid & 31;
  int n = lane & 15, mb = (lane < 16) ? 0 : 8;
  int m0 = wave * 16;
  __syncthreads();

  v16h aX = load_a16(&wih[m0 * 32], 32);
  v16h aH = load_a16(&whh[m0 * 32], 32);

  for (int s = 0; s < 5; ++s) {
    for (int i = tid; i < 512; i += 256) {
      int col = i >> 5, k = i & 31;
      xbT[col * 32 + k] =
          (k < 24) ? trajH[((size_t)s * NB_C + bc0 + col) * 24 + k] : (_Float16)0.0f;
    }
    __syncthreads();
    v8f acc = {};
    acc = __builtin_amdgcn_wmma_f32_16x16x32_f16(false, aX, false, load_bT(&xbT[0]),
                                                 (short)0, acc, false, false);
    acc = __builtin_amdgcn_wmma_f32_16x16x32_f16(false, aH, false, load_bT(&hbT[0]),
                                                 (short)0, acc, false, false);
#pragma unroll
    for (int r = 0; r < 8; ++r) { int m = m0 + mb + r; gates[m * GST + n] = acc[r] + bias[m]; }
    __syncthreads();
    for (int i = tid; i < 32 * 16; i += 256) {
      int u = i >> 4, col = i & 15;
      float ig = gates[u * GST + col], fg = gates[(32 + u) * GST + col];
      float gg = gates[(64 + u) * GST + col], og = gates[(96 + u) * GST + col];
      float c = sigf(fg) * cC[i] + sigf(ig) * tanhf(gg);
      float h = sigf(og) * tanhf(c);
      cC[i] = c; hC[i] = h;
      hbT[col * 32 + u] = (_Float16)h;
    }
    __syncthreads();
  }
  if (wave < 2) {   // diag = |h_last @ fw.T + fb| ; M padded 24->32
    int mt = wave * 16;
    v8f acc = {};
    v16h a = load_a16(&fwp[mt * 32], 32);
    acc = __builtin_amdgcn_wmma_f32_16x16x32_f16(false, a, false, load_bT(&hbT[0]),
                                                 (short)0, acc, false, false);
#pragma unroll
    for (int r = 0; r < 8; ++r) {
      int m = mt + mb + r;
      if (m < 24) Dout[(size_t)(bc0 + n) * 24 + m] = fabsf(acc[r] + fb_s[m]);
    }
  }
}

// =====================================================================
// Kernel 6: Kalman filter, one block per batch element, 5 scan steps.
// F = kron(I6, Fb) applied sparsely; solve S X = Pm by Gauss-Jordan
// (S symmetric => K = Pm S^-1, K[i][j] = X[j][i]).
// =====================================================================
__global__ void kf_kernel(const float* __restrict__ obs,
                          const float* __restrict__ trajF,
                          const float* __restrict__ Qd, const float* __restrict__ Rd,
                          float* __restrict__ out) {
  __shared__ float P[576], Pm[576], Aug[24 * 48];
  __shared__ float sv[24], sm[24], dz[24], fac[24];
  __shared__ float qd[24], rd[24];
  __shared__ float s_pivinv;
  int b = blockIdx.x, tid = threadIdx.x;
  if (tid < 24) {
    qd[tid] = Qd[(size_t)b * 24 + tid];
    rd[tid] = Rd[(size_t)b * 24 + tid];
    sv[tid] = (tid < 4) ? obs[((size_t)b * 100 + 99) * 400 + tid] : 0.0f;
  }
  for (int i = tid; i < 576; i += 256) P[i] = ((i / 24) == (i % 24)) ? 1.0f : 0.0f;
  __syncthreads();

  for (int s = 0; s < 5; ++s) {
    if (tid < 24) {                       // sm = F s
      float v = sv[tid];
      if ((tid & 3) < 2) v += DT_C * sv[tid + 2];
      sm[tid] = v;
    }
    for (int i = tid; i < 576; i += 256) {  // FP = F * P  (rows)
      int r = i / 24;
      float v = P[i];
      if ((r & 3) < 2) v += DT_C * P[i + 48];
      Aug[i] = v;
    }
    __syncthreads();
    for (int i = tid; i < 576; i += 256) {  // Pm = FP * F^T + Qdiag
      int r = i / 24, c = i % 24;
      float v = Aug[i];
      if ((c & 3) < 2) v += DT_C * Aug[i + 2];
      if (r == c) v += qd[r];
      Pm[i] = v;
    }
    __syncthreads();
    for (int i = tid; i < 1152; i += 256) { // Aug = [S | Pm], S = Pm + Rdiag
      int r = i / 48, c = i % 48;
      float v = (c < 24) ? Pm[r * 24 + c] : Pm[r * 24 + (c - 24)];
      if (c < 24 && r == c) v += rd[r];
      Aug[i] = v;
    }
    __syncthreads();
    for (int k = 0; k < 24; ++k) {          // Gauss-Jordan
      if (tid == 0) s_pivinv = 1.0f / Aug[k * 48 + k];
      __syncthreads();
      if (tid < 48) Aug[k * 48 + tid] *= s_pivinv;
      __syncthreads();
      if (tid < 24) fac[tid] = (tid == k) ? 0.0f : Aug[tid * 48 + k];
      __syncthreads();
      for (int i = tid; i < 1152; i += 256) {
        int r = i / 48;
        float f = fac[r];
        if (f != 0.0f) Aug[i] -= f * Aug[k * 48 + (i % 48)];
      }
      __syncthreads();
    }
    if (tid < 24) dz[tid] = trajF[((size_t)b * 5 + s) * 24 + tid] - sm[tid];
    __syncthreads();
    if (tid < 24) {                         // s_new = sm + K (z - sm)
      float acc = sm[tid];
      for (int j = 0; j < 24; ++j) acc += Aug[j * 48 + 24 + tid] * dz[j];
      sv[tid] = acc;
      out[((size_t)b * 5 + s) * 24 + tid] = acc;
    }
    for (int i = tid; i < 576; i += 256) {  // P_new = (I - K) Pm
      int r = i / 24, c = i % 24;
      float acc = Pm[i];
      for (int j = 0; j < 24; ++j) acc -= Aug[j * 48 + 24 + r] * Pm[j * 24 + c];
      P[i] = acc;
    }
    __syncthreads();
  }
}

// =====================================================================
// Launch
// =====================================================================
extern "C" void kernel_launch(void* const* d_in, const int* in_sizes, int n_in,
                              void* d_out, int out_size, void* d_ws, size_t ws_size,
                              hipStream_t stream) {
  const float* obs  = (const float*)d_in[0];
  const float* c1w  = (const float*)d_in[1];
  const float* c1b  = (const float*)d_in[2];
  const float* c2w  = (const float*)d_in[3];
  const float* c2b  = (const float*)d_in[4];
  const float* f1w  = (const float*)d_in[5];
  const float* f1b  = (const float*)d_in[6];
  const float* bn1g = (const float*)d_in[7];
  const float* bn1b = (const float*)d_in[8];
  const float* f2w  = (const float*)d_in[9];
  const float* f2b  = (const float*)d_in[10];
  const float* bn2g = (const float*)d_in[11];
  const float* bn2b = (const float*)d_in[12];
  const float* eWih = (const float*)d_in[13];
  const float* eWhh = (const float*)d_in[14];
  const float* ebih = (const float*)d_in[15];
  const float* ebhh = (const float*)d_in[16];
  const float* dWih = (const float*)d_in[17];
  const float* dWhh = (const float*)d_in[18];
  const float* dbih = (const float*)d_in[19];
  const float* dbhh = (const float*)d_in[20];
  const float* ow   = (const float*)d_in[21];
  const float* ob   = (const float*)d_in[22];
  const float* qWih = (const float*)d_in[23];
  const float* qWhh = (const float*)d_in[24];
  const float* qbih = (const float*)d_in[25];
  const float* qbhh = (const float*)d_in[26];
  const float* qfw  = (const float*)d_in[27];
  const float* qfb  = (const float*)d_in[28];
  const float* rWih = (const float*)d_in[29];
  const float* rWhh = (const float*)d_in[30];
  const float* rbih = (const float*)d_in[31];
  const float* rbhh = (const float*)d_in[32];
  const float* rfw  = (const float*)d_in[33];
  const float* rfb  = (const float*)d_in[34];

  char* ws = (char*)d_ws;
  _Float16* X1    = (_Float16*)(ws + 0);          // [51200][128] f16 = 13,107,200 B
  _Float16* featT = (_Float16*)(ws + 13107200);   // [100][512][64] f16 = 6,553,600 B
  float*    hfin  = (float*)   (ws + 19660800);   // 512*32 f32      =     65,536 B
  float*    cfin  = (float*)   (ws + 19726336);   // 512*32 f32      =     65,536 B
  float*    trajF = (float*)   (ws + 19791872);   // [512][5][24] f32 =   245,760 B
  _Float16* trajH = (_Float16*)(ws + 20037632);   // [5][512][24] f16 =   122,880 B
  float*    Qdiag = (float*)   (ws + 20160512);   // 512*24 f32      =     49,152 B
  float*    Rdiag = (float*)   (ws + 20209664);   // 512*24 f32      =     49,152 B

  conv_feat_kernel<<<NIMG_C / 8, 256, 0, stream>>>(obs, c1w, c1b, c2w, c2b, X1);
  fc_kernel<<<NIMG_C / 16, 256, 0, stream>>>(X1, f1w, f1b, bn1g, bn1b,
                                             f2w, f2b, bn2g, bn2b, featT);
  enc_lstm_kernel<<<NB_C / 16, 256, 0, stream>>>(featT, eWih, eWhh, ebih, ebhh, hfin, cfin);
  dec_kernel<<<NB_C / 16, 256, 0, stream>>>(obs, dWih, dWhh, dbih, dbhh, ow, ob,
                                            hfin, cfin, trajF, trajH);
  qr_lstm_kernel<<<NB_C / 16, 256, 0, stream>>>(trajH, qWih, qWhh, qbih, qbhh, qfw, qfb, Qdiag);
  qr_lstm_kernel<<<NB_C / 16, 256, 0, stream>>>(trajH, rWih, rWhh, rbih, rbhh, rfw, rfb, Rdiag);
  kf_kernel<<<NB_C, 256, 0, stream>>>(obs, trajF, Qdiag, Rdiag, (float*)d_out);
}
